// GRU_63934883168417
// MI455X (gfx1250) — matrix operand order repaired and checked by reading, not compiled, not run on hardware
//
#include <hip/hip_runtime.h>
#include <math.h>

typedef __attribute__((ext_vector_type(16))) _Float16 v16h;
typedef __attribute__((ext_vector_type(8)))  float    v8f;

#define WMMA_F16(a, b, c) \
  __builtin_amdgcn_wmma_f32_16x16x32_f16(false, (a), false, (b), (short)0, (c), false, false)

union U4H8 { uint4 u; _Float16 h[8]; };

__device__ __forceinline__ float sigmoidf_(float x) {
  return 1.0f / (1.0f + __expf(-x));
}

// Pack 16 consecutive f32 (four float4) into a v16h fragment (one-time paths).
__device__ __forceinline__ v16h pack_b16(float4 w0, float4 w1, float4 w2, float4 w3) {
  v16h b;
  b[0]  = (_Float16)w0.x; b[1]  = (_Float16)w0.y; b[2]  = (_Float16)w0.z; b[3]  = (_Float16)w0.w;
  b[4]  = (_Float16)w1.x; b[5]  = (_Float16)w1.y; b[6]  = (_Float16)w1.z; b[7]  = (_Float16)w1.w;
  b[8]  = (_Float16)w2.x; b[9]  = (_Float16)w2.y; b[10] = (_Float16)w2.z; b[11] = (_Float16)w2.w;
  b[12] = (_Float16)w3.x; b[13] = (_Float16)w3.y; b[14] = (_Float16)w3.z; b[15] = (_Float16)w3.w;
  return b;
}

// Build a v16h fragment from two aligned 16B chunks (halves j=0..7 / j=8..15).
__device__ __forceinline__ v16h pack_h16(uint4 lo, uint4 hi) {
  U4H8 ul, uh; ul.u = lo; uh.u = hi;
  v16h a;
#pragma unroll
  for (int j = 0; j < 8; ++j) { a[j] = ul.h[j]; a[j + 8] = uh.h[j]; }
  return a;
}

// ---------------------------------------------------------------------------
// Fused GRU layer. One block = 16 batch rows, one wave per 16 gate columns.
// Per step: hg+xg = [x_t | h] @ [W_ih | W_hh]^T via WMMA, K split at Ipad so
// the n-gate keeps separate x/h accumulators (n = tanh(xn + r*hn)).
// Weights live in VGPRs for all T steps; only the 16xKpad activation tile
// cycles through LDS. x_t staged via global_load_async_to_lds_b128 (f16 in).
// ---------------------------------------------------------------------------
template <bool XF32>
__global__ void __launch_bounds__(768, 1)
gru_layer_wmma(const void* __restrict__ xin,      // [B,T,I] f32 (layer1) or f16
               const float* __restrict__ w_ih,    // [3H, I]
               const float* __restrict__ w_hh,    // [3H, H]
               const float* __restrict__ b_ih,    // [3H]
               const float* __restrict__ b_hh,    // [3H]
               _Float16* __restrict__ yout,       // [B,T,H] = relu(h_t), f16
               int T, int I, int Ipad, int H)
{
  __shared__ _Float16 Alds[16][192];   // [m][k] : x_t cols [0,Ipad), h cols [Ipad,Ipad+H)
  __shared__ float    r_s[16][128];
  __shared__ float    z_s[16][128];

  const int b0     = blockIdx.x * 16;
  const int lane   = threadIdx.x & 31;
  const int wave   = threadIdx.x >> 5;
  const int khalf  = lane >> 4;
  const int l16    = lane & 15;
  const int Kpad   = Ipad + H;
  const int ksteps = Kpad >> 5;
  const int xsteps = Ipad >> 5;
  const int nIdx   = wave * 16 + l16;   // gate column in [0, 3H)

  // --- B fragments: fused [W_ih | W_hh] row nIdx, resident in VGPRs for all T ---
  v16h bfrag[6];
  for (int s = 0; s < ksteps; ++s) {
    v16h f;
    if (s < xsteps && I != Ipad) {
      // padded input projection (layer 1 only): guarded scalar loads, runs once
#pragma unroll
      for (int j = 0; j < 16; ++j) {
        int k = s * 32 + khalf * 16 + j;
        f[j] = (_Float16)((k < I) ? w_ih[(size_t)nIdx * I + k] : 0.0f);
      }
    } else {
      const float* src = (s < xsteps)
          ? (w_ih + (size_t)nIdx * I + s * 32 + khalf * 16)
          : (w_hh + (size_t)nIdx * H + (s * 32 - Ipad) + khalf * 16);
      const float4* p = (const float4*)src;
      f = pack_b16(p[0], p[1], p[2], p[3]);
    }
    bfrag[s] = f;
  }
  const float bi = b_ih[nIdx];
  const float bh = b_hh[nIdx];

  // zero h state + x padding columns
  for (int idx = threadIdx.x; idx < 16 * 192; idx += blockDim.x) {
    int m = idx / 192, k = idx % 192;
    if (k < Kpad) Alds[m][k] = (_Float16)0.0f;
  }
  __syncthreads();

  for (int t = 0; t < T; ++t) {
    // ---- phase 1: stage x_t tile into LDS ----
    if constexpr (XF32) {
      for (int idx = threadIdx.x; idx < 16 * I; idx += blockDim.x) {
        int m = idx / I, i = idx % I;
        size_t off = ((size_t)(b0 + m) * T + t) * (size_t)I + i;
        Alds[m][i] = (_Float16)((const float*)xin)[off];
      }
    } else {
      // async DMA: 16B per lane, ASYNCcnt-tracked, no VGPR staging
      const _Float16* xf = (const _Float16*)xin;
      const int cI = I >> 3;                       // 16B chunks per row
      for (int idx = threadIdx.x; idx < 16 * cI; idx += blockDim.x) {
        int m = idx / cI, i = (idx % cI) << 3;
        unsigned long long src =
            (unsigned long long)(xf + ((size_t)(b0 + m) * T + t) * (size_t)I + i);
        unsigned dst = (unsigned)(size_t)&Alds[m][i];
        asm volatile("global_load_async_to_lds_b128 %0, %1, off"
                     :: "v"(dst), "v"(src) : "memory");
      }
      asm volatile("s_wait_asynccnt 0x0" ::: "memory");
    }
    if (threadIdx.x < 16 && t + 1 < T) {           // global_prefetch_b8 next step
      size_t off = ((size_t)(b0 + threadIdx.x) * T + (t + 1)) * (size_t)I;
      __builtin_prefetch(XF32 ? (const void*)((const float*)xin + off)
                              : (const void*)((const _Float16*)xin + off), 0, 1);
    }
    __syncthreads();

    // ---- phase 2: WMMA over fused K, accumulators split at the x|h boundary ----
    v8f accx = {};
    v8f acch = {};
    for (int s = 0; s < ksteps; ++s) {
      uint4 lo = *(const uint4*)&Alds[l16][s * 32 + khalf * 16];        // ds_load_b128
      uint4 hi = *(const uint4*)&Alds[l16][s * 32 + khalf * 16 + 8];   // ds_load_b128
      v16h a = pack_h16(lo, hi);
      if (s < xsteps) accx = WMMA_F16(a, bfrag[s], accx);
      else            acch = WMMA_F16(a, bfrag[s], acch);
    }

    if (nIdx < 2 * H) {   // whole-wave uniform: r and z gate tiles
#pragma unroll
      for (int v = 0; v < 8; ++v) {
        int m = v + 8 * khalf;
        float g = sigmoidf_(accx[v] + acch[v] + bi + bh);
        if (nIdx < H) r_s[m][nIdx] = g;
        else          z_s[m][nIdx - H] = g;
      }
    }
    __syncthreads();

    // ---- phase 3: n-gate waves finish the state update ----
    if (nIdx >= 2 * H) {
      int c = nIdx - 2 * H;
#pragma unroll
      for (int v = 0; v < 8; ++v) {
        int m  = v + 8 * khalf;
        float r  = r_s[m][c];
        float z  = z_s[m][c];
        float n  = tanhf(accx[v] + bi + r * (acch[v] + bh));
        float hp = (float)Alds[m][Ipad + c];
        float hn = (1.0f - z) * n + z * hp;
        Alds[m][Ipad + c] = (_Float16)hn;          // recurrent state (pre-ReLU)
        yout[((size_t)(b0 + m) * T + t) * (size_t)H + c] = (_Float16)fmaxf(hn, 0.0f);
      }
    }
    __syncthreads();
  }
}

// ---------------------------------------------------------------------------
// One-shot weight conversion f32 -> f16 (keeps dense hot loops cvt-free).
// ---------------------------------------------------------------------------
__global__ void f32_to_f16_kernel(const float* __restrict__ src,
                                  _Float16* __restrict__ dst, int n4) {
  int i = (blockIdx.x * blockDim.x + threadIdx.x);
  if (i < n4) {
    float4 v = ((const float4*)src)[i];
    U4H8 o;
    o.h[0] = (_Float16)v.x; o.h[1] = (_Float16)v.y;
    o.h[2] = (_Float16)v.z; o.h[3] = (_Float16)v.w;
    ((uint2*)dst)[i] = make_uint2(o.u.x, o.u.y);
  }
}

// ---------------------------------------------------------------------------
// Split-K dense1: each block computes an f32 partial over a K-slice.
// part[nsplit, M, N]; deterministic (no atomics). K/32 must divide by nsplit.
// ---------------------------------------------------------------------------
__global__ void __launch_bounds__(512, 1)
dense1_partial_wmma(const _Float16* __restrict__ A,   // [M,K] f16
                    const _Float16* __restrict__ W,   // [N,K] f16
                    float* __restrict__ part,         // [nsplit,M,N]
                    int M, int N, int K, int ntN, int nsplit)
{
  const int lane   = threadIdx.x & 31;
  const int wave   = threadIdx.x >> 5;
  const int khalf  = lane >> 4;
  const int l16    = lane & 15;
  const int nwaves = blockDim.x >> 5;
  const int mtBlk  = nwaves / ntN;
  const int slice  = blockIdx.x % nsplit;
  const int mblk   = blockIdx.x / nsplit;
  const int mt     = wave / ntN;
  const int nt     = wave % ntN;
  const int row0   = (mblk * mtBlk + mt) * 16;
  const int n      = nt * 16 + l16;
  const int sliceSteps = (K >> 5) / nsplit;
  const int s0 = slice * sliceSteps, s1 = s0 + sliceSteps;

  const _Float16* arow = A + (size_t)(row0 + l16) * (size_t)K;
  const _Float16* wrow = W + (size_t)n * (size_t)K;

  v8f acc = {};
  for (int s = s0; s < s1; ++s) {
    uint4 alo = *(const uint4*)(arow + s * 32 + khalf * 16);
    uint4 ahi = *(const uint4*)(arow + s * 32 + khalf * 16 + 8);
    uint4 blo = *(const uint4*)(wrow + s * 32 + khalf * 16);
    uint4 bhi = *(const uint4*)(wrow + s * 32 + khalf * 16 + 8);
    if (s + 8 < s1) {
      __builtin_prefetch(arow + (s + 8) * 32, 0, 1);
      __builtin_prefetch(wrow + (s + 8) * 32, 0, 1);
    }
    acc = WMMA_F16(pack_h16(alo, ahi), pack_h16(blo, bhi), acc);
  }

  float* dst = part + ((size_t)slice * M + row0) * (size_t)N + n;
#pragma unroll
  for (int v = 0; v < 8; ++v) {
    int m = v + 8 * khalf;
    dst[(size_t)m * N] = acc[v];
  }
}

__global__ void dense1_reduce_kernel(const float* __restrict__ part,
                                     const float* __restrict__ bias,
                                     _Float16* __restrict__ out,
                                     int M, int N, int nsplit) {
  int i = blockIdx.x * blockDim.x + threadIdx.x;
  if (i >= M * N) return;
  float s = bias[i % N];
  for (int p = 0; p < nsplit; ++p) s += part[(size_t)p * M * N + i];
  out[i] = (_Float16)fmaxf(s, 0.0f);
}

// ---------------------------------------------------------------------------
// Generic WMMA dense layer (f16 weights): out = act(A @ W^T + bias).
// Wave w -> (m-tile, n-tile) = (w / ntN, w % ntN). K multiple of 32.
// Out-of-range N columns (dense3 pad 250->256) clamp the W row pointer and
// never store: column n of C depends only on row n of W.
// ---------------------------------------------------------------------------
__global__ void __launch_bounds__(512, 1)
dense_wmma(const _Float16* __restrict__ A,   // [M,K] f16
           const _Float16* __restrict__ W,   // [N,K] f16
           const float* __restrict__ bias,   // [N]
           void* __restrict__ out,           // [M,N] f16 or f32
           int M, int N, int K, int ntN, int relu, int out_f32)
{
  const int lane   = threadIdx.x & 31;
  const int wave   = threadIdx.x >> 5;
  const int khalf  = lane >> 4;
  const int l16    = lane & 15;
  const int nwaves = blockDim.x >> 5;
  const int mtBlk  = nwaves / ntN;
  const int mt     = wave / ntN;
  const int nt     = wave % ntN;
  const int row0   = (blockIdx.x * mtBlk + mt) * 16;
  const int n      = nt * 16 + l16;
  const int nc     = (n < N) ? n : (N - 1);
  const int ksteps = K >> 5;

  const _Float16* arow = A + (size_t)(row0 + l16) * (size_t)K;
  const _Float16* wrow = W + (size_t)nc * (size_t)K;

  v8f acc = {};
  for (int s = 0; s < ksteps; ++s) {
    uint4 alo = *(const uint4*)(arow + s * 32 + khalf * 16);
    uint4 ahi = *(const uint4*)(arow + s * 32 + khalf * 16 + 8);
    uint4 blo = *(const uint4*)(wrow + s * 32 + khalf * 16);
    uint4 bhi = *(const uint4*)(wrow + s * 32 + khalf * 16 + 8);
    acc = WMMA_F16(pack_h16(alo, ahi), pack_h16(blo, bhi), acc);
  }

  const float bv = bias[nc];
#pragma unroll
  for (int v = 0; v < 8; ++v) {
    int m = v + 8 * khalf;
    float val = acc[v] + bv;
    if (relu) val = fmaxf(val, 0.0f);
    if (n < N) {
      size_t off = (size_t)(row0 + m) * (size_t)N + n;
      if (out_f32) ((float*)out)[off] = val;
      else         ((_Float16*)out)[off] = (_Float16)val;
    }
  }
  (void)M;
}

extern "C" void kernel_launch(void* const* d_in, const int* in_sizes, int n_in,
                              void* d_out, int out_size, void* d_ws, size_t ws_size,
                              hipStream_t stream) {
  (void)in_sizes; (void)n_in; (void)out_size; (void)ws_size;
  const int B = 2048, T = 543;
  const int K1 = T * 64;           // 34752, 1086 k-steps = 6 * 181
  const int NSPLIT = 6;

  const float* x     = (const float*)d_in[0];
  const float* w_ih1 = (const float*)d_in[1];
  const float* w_hh1 = (const float*)d_in[2];
  const float* b_ih1 = (const float*)d_in[3];
  const float* b_hh1 = (const float*)d_in[4];
  const float* w_ih2 = (const float*)d_in[5];
  const float* w_hh2 = (const float*)d_in[6];
  const float* b_ih2 = (const float*)d_in[7];
  const float* b_hh2 = (const float*)d_in[8];
  const float* w_ih3 = (const float*)d_in[9];
  const float* w_hh3 = (const float*)d_in[10];
  const float* b_ih3 = (const float*)d_in[11];
  const float* b_hh3 = (const float*)d_in[12];
  const float* w1    = (const float*)d_in[13];
  const float* bd1   = (const float*)d_in[14];
  const float* w2    = (const float*)d_in[15];
  const float* bd2   = (const float*)d_in[16];
  const float* w3    = (const float*)d_in[17];
  const float* bd3   = (const float*)d_in[18];

  // ---- workspace layout (256B aligned slots; y3 reuses y1's slot) ----
  char* ws = (char*)d_ws;
  size_t off = 0;
  auto alloc = [&](size_t bytes) {
    char* p = ws + off;
    off += (bytes + 255) & ~(size_t)255;
    return p;
  };
  _Float16* y1     = (_Float16*)alloc((size_t)B * T * 64 * 2);    // 142 MB (also y3)
  _Float16* y2     = (_Float16*)alloc((size_t)B * T * 128 * 2);   // 285 MB
  _Float16* d1o    = (_Float16*)alloc((size_t)B * 64 * 2);
  _Float16* d2o    = (_Float16*)alloc((size_t)B * 32 * 2);
  _Float16* w1h    = (_Float16*)alloc((size_t)64 * K1 * 2);       // 4.45 MB
  _Float16* w2h    = (_Float16*)alloc((size_t)32 * 64 * 2);
  _Float16* w3h    = (_Float16*)alloc((size_t)250 * 32 * 2);
  float*    d1part = (float*)alloc((size_t)NSPLIT * B * 64 * 4);  // 3.1 MB

  // ---- one-shot weight conversions (overlap with GRU work on the stream) ----
  f32_to_f16_kernel<<<(64 * K1 / 4 + 255) / 256, 256, 0, stream>>>(w1, w1h, 64 * K1 / 4);
  f32_to_f16_kernel<<<(32 * 64 / 4 + 255) / 256, 256, 0, stream>>>(w2, w2h, 32 * 64 / 4);
  f32_to_f16_kernel<<<(250 * 32 / 4 + 255) / 256, 256, 0, stream>>>(w3, w3h, 250 * 32 / 4);

  const dim3 gGru(B / 16);
  // layer 1: I=3 (pad 32), H=64  -> 3H=192 -> 12 waves
  gru_layer_wmma<true ><<<gGru, 12 * 32, 0, stream>>>(x,  w_ih1, w_hh1, b_ih1, b_hh1, y1, T, 3,   32,  64);
  // layer 2: I=64, H=128 -> 3H=384 -> 24 waves
  gru_layer_wmma<false><<<gGru, 24 * 32, 0, stream>>>(y1, w_ih2, w_hh2, b_ih2, b_hh2, y2, T, 64,  64,  128);
  // layer 3: I=128, H=64 -> 12 waves (output reuses y1's buffer)
  gru_layer_wmma<false><<<gGru, 12 * 32, 0, stream>>>(y2, w_ih3, w_hh3, b_ih3, b_hh3, y1, T, 128, 128, 64);

  // dense1 split-K: 32 m-blocks x 6 K-slices = 192 blocks to saturate HBM
  dense1_partial_wmma<<<(B / 64) * NSPLIT, 512, 0, stream>>>(y1, w1h, d1part, B, 64, K1, 4, NSPLIT);
  dense1_reduce_kernel<<<(B * 64 + 255) / 256, 256, 0, stream>>>(d1part, bd1, d1o, B, 64, NSPLIT);
  // dense2: N=32 -> ntN=2, 128 rows/block
  dense_wmma<<<B / 128, 512, 0, stream>>>(d1o, w2h, bd2, d2o,  B, 32,  64, 2,  1, 0);
  // dense3: N=250 (pad 256) -> ntN=16, 16 rows/block, f32 output
  dense_wmma<<<B / 16, 512, 0, stream>>>(d2o, w3h, bd3, d_out, B, 250, 32, 16, 1, 1);
}
